// EncoderBlock_82970178224309
// MI455X (gfx1250) — matrix operand order repaired and checked by reading, not compiled, not run on hardware
//
#include <hip/hip_runtime.h>
#include <hip/hip_bf16.h>

typedef __attribute__((ext_vector_type(16))) __bf16 v16bf;
typedef __attribute__((ext_vector_type(8)))  __bf16 v8bf;
typedef __attribute__((ext_vector_type(8)))  float  v8f;
typedef __attribute__((ext_vector_type(4)))  unsigned int u32x4;
typedef __attribute__((ext_vector_type(8)))  int i32x8;
typedef __attribute__((ext_vector_type(4)))  int i32x4;

#define B_    8
#define L_    512
#define D_    1024
#define H_    16
#define DH_   64
#define NREL  129     // 2*MAXD+1
#define NRELP 160     // padded to multiple of 32 for WMMA K
#define MTOK  (B_*L_) // 4096
#define BH_   (B_*H_) // 128

// ---------- bf16 helpers ----------
static __device__ __forceinline__ unsigned short f2bf(float f) {
  unsigned u = __float_as_uint(f);
  unsigned r = (u + 0x7FFFu + ((u >> 16) & 1u)) >> 16; // RNE
  return (unsigned short)r;
}
static __device__ __forceinline__ float bf2f(unsigned short h) {
  return __uint_as_float(((unsigned)h) << 16);
}
// A-operand fragment: lane half g holds K in [16g,16g+16) (contiguous k order, same as the B loads)
static __device__ __forceinline__ v16bf a_frag(const unsigned short* p, int lg) {
  v8bf lo = *reinterpret_cast<const v8bf*>(p + 16 * lg);
  v8bf hi = *reinterpret_cast<const v8bf*>(p + 16 * lg + 8);
  return __builtin_shufflevector(lo, hi, 0,1,2,3,4,5,6,7,8,9,10,11,12,13,14,15);
}
static __device__ __forceinline__ v8f wmma_bf16(v16bf a, v16bf b, v8f c) {
  return __builtin_amdgcn_wmma_f32_16x16x32_bf16(false, a, false, b, (short)0, c, false, false);
}

// ---------- Tensor Data Mover: 2D tile (rows x row_elems bf16) global -> LDS ----------
// D# per CDNA5 ISA: group0 = {count/lds/global/type},
// group1 = {data_size, tensor dims, tile dims, dim0 stride}.
static __device__ __forceinline__ unsigned lds_off(const void* p) {
  return (unsigned)(unsigned long long)p;
}
static __device__ __forceinline__ void tdm_load_2d(unsigned lds_addr,
                                                   const void* gptr,
                                                   unsigned rows,
                                                   unsigned row_elems,
                                                   unsigned row_stride_elems) {
  unsigned long long ga = (unsigned long long)gptr;
  const unsigned td1 = 0x00100000u;          // huge tensor_dim1: no OOB clamp
  u32x4 g0;
  g0[0] = 1u;                                // count=1, user descriptor
  g0[1] = lds_addr;                          // lds_addr  (bits 63:32)
  g0[2] = (unsigned)(ga & 0xffffffffu);      // global_addr lo (bits 95:64)
  g0[3] = (unsigned)((ga >> 32) & 0x01ffffffu) | (2u << 30); // addr hi | type=2
  i32x8 g1;
  g1[0] = (int)(1u << 16);                   // data_size=1 -> 2 bytes/elem
  g1[1] = (int)((row_stride_elems & 0xffffu) << 16);          // tensor_dim0 lo16
  g1[2] = (int)((row_stride_elems >> 16) | ((td1 & 0xffffu) << 16)); // td0 hi | td1 lo
  g1[3] = (int)((td1 >> 16) | (row_elems << 16));             // td1 hi | tile_dim0
  g1[4] = (int)(rows & 0xffffu);             // tile_dim1 (tile_dim2 = 0)
  g1[5] = (int)row_stride_elems;             // tensor_dim0_stride lo32
  g1[6] = 0;
  g1[7] = 0;
  i32x4 z4 = {0, 0, 0, 0};
#if defined(__clang_major__) && (__clang_major__ >= 23)
  i32x8 z8 = {0, 0, 0, 0, 0, 0, 0, 0};
  __builtin_amdgcn_tensor_load_to_lds(g0, g1, z4, z4, z8, 0);
#else
  __builtin_amdgcn_tensor_load_to_lds(g0, g1, z4, z4, 0);
#endif
}

// ---------- packing kernels ----------
__global__ void pack_bf16_kernel(const float* __restrict__ src,
                                 unsigned short* __restrict__ dst, int n) {
  int i = blockIdx.x * blockDim.x + threadIdx.x;
  if (i < n) dst[i] = f2bf(src[i]);
}
// W is (K=D, N=D) row-major; pack as Wp[n][k] (N-major, K contiguous) for B-operand loads
__global__ void pack_wt_kernel(const float* __restrict__ W,
                               unsigned short* __restrict__ Wp) {
  int i = blockIdx.x * blockDim.x + threadIdx.x; // D_*D_ elements
  int n = i >> 10, k = i & 1023;
  Wp[i] = f2bf(W[k * D_ + n]);
}
// posV (NREL, DH) -> posVp[d][r] padded to NRELP in r (B-operand for Ap @ posV)
__global__ void pack_posv_kernel(const float* __restrict__ posV,
                                 unsigned short* __restrict__ posVp) {
  int i = blockIdx.x * blockDim.x + threadIdx.x;
  if (i >= DH_ * NRELP) return;
  int n = i / NRELP, k = i % NRELP;
  posVp[i] = (k < NREL) ? f2bf(posV[k * DH_ + n]) : (unsigned short)0;
}

// ---------- bf16 WMMA GEMM with TDM double-buffered LDS staging ----------
// C = A(MxK) @ Wp(NxK packed) + bias.  Block tile 64x128, 4 waves of 32x64.
// mode 0: row-major out (f32 and/or bf16). mode 1: bf16 out at (B,H,L,DH).
// mode 2: bf16 out transposed (B,H,DH,L)  [used for V].
__global__ __launch_bounds__(128) void gemm_bf16_kernel(
    const unsigned short* __restrict__ A,
    const unsigned short* __restrict__ Wp,
    const float* __restrict__ bias,
    float* __restrict__ outF,
    unsigned short* __restrict__ outB,
    int M, int N, int K, int mode, int relu)
{
  __shared__ unsigned short At[2][64 * 32];   // 2 x 4 KB
  __shared__ unsigned short Bt[2][128 * 32];  // 2 x 8 KB

  const int wave = threadIdx.x >> 5;
  const int lane = threadIdx.x & 31;
  const int lm = lane & 15, lg = lane >> 4;
  const int wm = wave & 1, wn = wave >> 1;
  const int mblk = blockIdx.x * 64;
  const int nblk = blockIdx.y * 128;

  v8f zero = {};
  v8f acc[2][4];
#pragma unroll
  for (int i = 0; i < 2; ++i)
#pragma unroll
    for (int j = 0; j < 4; ++j) acc[i][j] = zero;

  const bool issuer = (wave == 0);
  const int ksteps = K >> 5;

  if (issuer) {
    tdm_load_2d(lds_off(&At[0][0]), A + (size_t)mblk * K, 64, 32, K);
    tdm_load_2d(lds_off(&Bt[0][0]), Wp + (size_t)nblk * K, 128, 32, K);
  }

  for (int kt = 0; kt < ksteps; ++kt) {
    const int cur = kt & 1, nxt = cur ^ 1;
    __syncthreads(); // buf[nxt] no longer read by any wave
    if (issuer) {
      if (kt + 1 < ksteps) {
        tdm_load_2d(lds_off(&At[nxt][0]), A + (size_t)mblk * K + (kt + 1) * 32, 64, 32, K);
        tdm_load_2d(lds_off(&Bt[nxt][0]), Wp + (size_t)nblk * K + (kt + 1) * 32, 128, 32, K);
        __builtin_amdgcn_s_wait_tensorcnt(2); // kt's two loads complete (in-order)
      } else {
        __builtin_amdgcn_s_wait_tensorcnt(0);
      }
    }
    __syncthreads(); // buf[cur] visible to all waves

    v16bf a0 = a_frag(&At[cur][(wm * 32 + lm) * 32], lg);
    v16bf a1 = a_frag(&At[cur][(wm * 32 + 16 + lm) * 32], lg);
#pragma unroll
    for (int nt = 0; nt < 4; ++nt) {
      v16bf bmat = *reinterpret_cast<const v16bf*>(&Bt[cur][(wn * 64 + nt * 16 + lm) * 32 + 16 * lg]);
      acc[0][nt] = wmma_bf16(a0, bmat, acc[0][nt]);
      acc[1][nt] = wmma_bf16(a1, bmat, acc[1][nt]);
    }
  }

  const int m0 = mblk + wm * 32;
  const int n0 = nblk + wn * 64;
#pragma unroll
  for (int mt = 0; mt < 2; ++mt) {
#pragma unroll
    for (int nt = 0; nt < 4; ++nt) {
      int nn = n0 + nt * 16 + lm;
      float bb = bias ? bias[nn] : 0.0f;
#pragma unroll
      for (int r = 0; r < 8; ++r) {
        int mm = m0 + mt * 16 + r + 8 * lg; // C layout: M = r + 8*(lane>=16)
        float v = acc[mt][nt][r] + bb;
        if (relu) v = fmaxf(v, 0.0f);
        if (mode == 0) {
          size_t idx = (size_t)mm * N + nn;
          if (outF) outF[idx] = v;
          if (outB) outB[idx] = f2bf(v);
        } else {
          int bidx = mm >> 9, lidx = mm & 511;
          int hidx = nn >> 6, didx = nn & 63;
          size_t idx;
          if (mode == 1) idx = ((size_t)((bidx * H_ + hidx) * L_ + lidx)) * DH_ + didx;
          else           idx = ((size_t)((bidx * H_ + hidx) * DH_ + didx)) * L_ + lidx;
          outB[idx] = f2bf(v);
        }
      }
    }
  }
}

// ---------- Qp[row, r] = Q[row,:] . posK[r,:]  (row = (b*H+h)*L + i) ----------
__global__ __launch_bounds__(160) void qp_kernel(
    const unsigned short* __restrict__ Qbf,
    const float* __restrict__ posK,
    float* __restrict__ Qp)
{
  __shared__ float q[DH_];
  size_t row = blockIdx.x;
  if (threadIdx.x < DH_) q[threadIdx.x] = bf2f(Qbf[row * DH_ + threadIdx.x]);
  __syncthreads();
  int r = threadIdx.x;
  if (r < NREL) {
    float s = 0.f;
#pragma unroll 8
    for (int d = 0; d < DH_; ++d) s += q[d] * posK[r * DH_ + d];
    Qp[row * NREL + r] = s;
  }
}

// ---------- attention: per (b,h) and 16-query-row block ----------
__global__ __launch_bounds__(128) void attn_kernel(
    const unsigned short* __restrict__ Qbf,   // (B,H,L,DH) bf16
    const unsigned short* __restrict__ Kbf,   // (B,H,L,DH) bf16
    const unsigned short* __restrict__ Vt,    // (B,H,DH,L) bf16
    const float* __restrict__ Qp,             // (B*H*L, NREL)
    const unsigned short* __restrict__ posVp, // (DH, NRELP) bf16
    const float* __restrict__ pad_mask,       // (B,L)
    unsigned short* __restrict__ attnIn)      // (B,L,D) bf16
{
  __shared__ float          S[16][512];        // 32 KB
  __shared__ unsigned short Abf[16][512];      // 16 KB
  __shared__ float          Apf[16][NRELP];    // 10 KB
  __shared__ unsigned short Apbf[16][NRELP];   //  5 KB
  __shared__ float          red[16][8];
  __shared__ float          rowmax[16];
  __shared__ float          rowsum[16];

  const int tid = threadIdx.x;
  const int wave = tid >> 5, lane = tid & 31;
  const int lm = lane & 15, lg = lane >> 4;
  const int bh = blockIdx.y;
  const int b = bh >> 4, h = bh & 15;
  const int i0 = blockIdx.x * 16;

  for (int t = tid; t < 16 * NRELP; t += 128) (&Apf[0][0])[t] = 0.0f;

  // ---- pass 1: S = (Q K^T + gathered pos-score) * 1/8 * mask
  const unsigned short* Qrow = Qbf + (size_t)(bh * L_ + i0 + lm) * DH_;
  v16bf aq0 = a_frag(Qrow, lg);
  v16bf aq1 = a_frag(Qrow + 32, lg);
  for (int jt = wave; jt < 32; jt += 4) {
    int j0 = jt * 16;
    v8f acc = {};
    const unsigned short* Krow = Kbf + (size_t)(bh * L_ + j0 + lm) * DH_ + 16 * lg;
    v16bf kb0 = *reinterpret_cast<const v16bf*>(Krow);
    v16bf kb1 = *reinterpret_cast<const v16bf*>(Krow + 32);
    acc = wmma_bf16(aq0, kb0, acc);
    acc = wmma_bf16(aq1, kb1, acc);
    int jj = j0 + lm;
    float mk = pad_mask[b * L_ + jj];
#pragma unroll
    for (int r = 0; r < 8; ++r) {
      int il = r + 8 * lg;
      int i = i0 + il;
      int rel = jj - i; rel = rel < -64 ? -64 : (rel > 64 ? 64 : rel); rel += 64;
      float v = (acc[r] + Qp[(size_t)(bh * L_ + i) * NREL + rel]) * 0.125f * mk;
      S[il][jj] = v;
    }
  }
  __syncthreads();

  // ---- softmax over each row of S (16 rows x 512 cols), 8 segments per row
  const int row = tid >> 3, seg = tid & 7;
  {
    float mx = -3.4e38f;
    for (int c = seg * 64; c < seg * 64 + 64; ++c) mx = fmaxf(mx, S[row][c]);
    red[row][seg] = mx;
  }
  __syncthreads();
  if (seg == 0) {
    float m = red[row][0];
#pragma unroll
    for (int k = 1; k < 8; ++k) m = fmaxf(m, red[row][k]);
    rowmax[row] = m;
  }
  __syncthreads();
  {
    float m = rowmax[row], s = 0.f;
    for (int c = seg * 64; c < seg * 64 + 64; ++c) {
      float e = __expf(S[row][c] - m);
      S[row][c] = e;
      s += e;
    }
    red[row][seg] = s;
  }
  __syncthreads();
  if (seg == 0) {
    float s = 0.f;
#pragma unroll
    for (int k = 0; k < 8; ++k) s += red[row][k];
    rowsum[row] = s;
  }
  __syncthreads();
  {
    float inv = 1.0f / rowsum[row];
    int i = i0 + row;
    for (int c = seg * 64; c < seg * 64 + 64; ++c) {
      float a = S[row][c] * inv;
      Abf[row][c] = f2bf(a);
      int rel = c - i; rel = rel < -64 ? -64 : (rel > 64 ? 64 : rel); rel += 64;
      atomicAdd(&Apf[row][rel], a);   // ds_add_f32
    }
  }
  __syncthreads();
  for (int t = tid; t < 16 * NRELP; t += 128)
    (&Apbf[0][0])[t] = f2bf((&Apf[0][0])[t]);
  __syncthreads();

  // ---- pass 2: O = A @ V + Ap @ posV   (each wave owns one 16-wide d-tile)
  const int d0 = wave * 16;
  v8f o = {};
  const unsigned short* Vrow = Vt + (size_t)(bh * DH_ + d0 + lm) * L_ + 16 * lg;
  for (int kt = 0; kt < 16; ++kt) {
    v16bf a = a_frag(&Abf[lm][kt * 32], lg);
    v16bf bv = *reinterpret_cast<const v16bf*>(Vrow + kt * 32);
    o = wmma_bf16(a, bv, o);
  }
  const unsigned short* Prow = posVp + (size_t)(d0 + lm) * NRELP + 16 * lg;
#pragma unroll
  for (int kt = 0; kt < 5; ++kt) {
    v16bf a = a_frag(&Apbf[lm][kt * 32], lg);
    v16bf bv = *reinterpret_cast<const v16bf*>(Prow + kt * 32);
    o = wmma_bf16(a, bv, o);
  }
#pragma unroll
  for (int r = 0; r < 8; ++r) {
    int i = i0 + r + 8 * lg;
    int d = d0 + lm;
    attnIn[(size_t)(b * L_ + i) * D_ + h * DH_ + d] = f2bf(o[r]);
  }
}

// ---------- BatchNorm stats over (B,L): mean/var per channel ----------
__global__ void bn_stats_kernel(const float* __restrict__ y,
                                float* __restrict__ mean, float* __restrict__ var) {
  int c = blockIdx.x * blockDim.x + threadIdx.x;
  if (c >= D_) return;
  float s = 0.f, ss = 0.f;
  for (int m = 0; m < MTOK; ++m) {
    float v = y[(size_t)m * D_ + c];
    s += v; ss += v * v;
  }
  float mu = s * (1.0f / MTOK);
  mean[c] = mu;
  var[c] = ss * (1.0f / MTOK) - mu * mu;
}

// ---------- BN apply + residual (+optional ReLU, optional bf16 copy) ----------
__global__ void bn_apply_kernel(const float* __restrict__ y,
                                const float* __restrict__ mean,
                                const float* __restrict__ var,
                                const float* __restrict__ gamma,
                                const float* __restrict__ beta,
                                const float* __restrict__ resid,
                                float* __restrict__ outF,
                                unsigned short* __restrict__ outB,
                                int relu) {
  int idx = blockIdx.x * blockDim.x + threadIdx.x;
  if (idx >= MTOK * D_) return;
  int c = idx & (D_ - 1);
  float v = (y[idx] - mean[c]) * rsqrtf(var[c] + 1e-5f) * gamma[c] + beta[c] + resid[idx];
  if (relu) v = fmaxf(v, 0.0f);
  outF[idx] = v;
  if (outB) outB[idx] = f2bf(v);
}

// ---------- host orchestration ----------
extern "C" void kernel_launch(void* const* d_in, const int* in_sizes, int n_in,
                              void* d_out, int out_size, void* d_ws, size_t ws_size,
                              hipStream_t stream) {
  (void)in_sizes; (void)n_in; (void)out_size; (void)ws_size;
  const float* x    = (const float*)d_in[0];
  const float* pm   = (const float*)d_in[1];
  const float* Wq   = (const float*)d_in[2];
  const float* bq   = (const float*)d_in[3];
  const float* Wk   = (const float*)d_in[4];
  const float* bk   = (const float*)d_in[5];
  const float* Wv   = (const float*)d_in[6];
  const float* bv   = (const float*)d_in[7];
  const float* Wo   = (const float*)d_in[8];
  const float* bo   = (const float*)d_in[9];
  const float* posK = (const float*)d_in[10];
  const float* posV = (const float*)d_in[11];
  const float* ga   = (const float*)d_in[12];
  const float* bta  = (const float*)d_in[13];
  const float* W1   = (const float*)d_in[14];
  const float* b1   = (const float*)d_in[15];
  const float* W2   = (const float*)d_in[16];
  const float* b2   = (const float*)d_in[17];
  const float* gf   = (const float*)d_in[18];
  const float* bfv  = (const float*)d_in[19];
  float* out = (float*)d_out;

  char* ws = (char*)d_ws;
  size_t off = 0;
  auto alloc = [&](size_t bytes) -> char* {
    char* p = ws + off;
    off += (bytes + 255) & ~(size_t)255;
    return p;
  };
  unsigned short* xb    = (unsigned short*)alloc((size_t)MTOK * D_ * 2);
  unsigned short* Wqp   = (unsigned short*)alloc((size_t)D_ * D_ * 2);
  unsigned short* Wkp   = (unsigned short*)alloc((size_t)D_ * D_ * 2);
  unsigned short* Wvp   = (unsigned short*)alloc((size_t)D_ * D_ * 2);
  unsigned short* Wop   = (unsigned short*)alloc((size_t)D_ * D_ * 2);
  unsigned short* W1p   = (unsigned short*)alloc((size_t)D_ * D_ * 2);
  unsigned short* W2p   = (unsigned short*)alloc((size_t)D_ * D_ * 2);
  unsigned short* posVp = (unsigned short*)alloc((size_t)DH_ * NRELP * 2);
  unsigned short* Qbf   = (unsigned short*)alloc((size_t)BH_ * L_ * DH_ * 2);
  unsigned short* Kbf   = (unsigned short*)alloc((size_t)BH_ * L_ * DH_ * 2);
  unsigned short* Vt    = (unsigned short*)alloc((size_t)BH_ * DH_ * L_ * 2);
  float*          Qp    = (float*)alloc((size_t)BH_ * L_ * NREL * 4);
  unsigned short* attnIn= (unsigned short*)alloc((size_t)MTOK * D_ * 2);
  float*          attnF = (float*)alloc((size_t)MTOK * D_ * 4);
  float*          x1F   = (float*)alloc((size_t)MTOK * D_ * 4);
  unsigned short* x1B   = (unsigned short*)alloc((size_t)MTOK * D_ * 2);
  unsigned short* h1B   = (unsigned short*)alloc((size_t)MTOK * D_ * 2);
  float*          ffF   = (float*)alloc((size_t)MTOK * D_ * 4);
  float*          mean1 = (float*)alloc(D_ * 4);
  float*          var1  = (float*)alloc(D_ * 4);
  float*          mean2 = (float*)alloc(D_ * 4);
  float*          var2  = (float*)alloc(D_ * 4);

  const int NTOT = MTOK * D_;

  // pack activations + weights to bf16 layouts
  pack_bf16_kernel<<<(NTOT + 255) / 256, 256, 0, stream>>>(x, xb, NTOT);
  pack_wt_kernel<<<(D_ * D_) / 256, 256, 0, stream>>>(Wq, Wqp);
  pack_wt_kernel<<<(D_ * D_) / 256, 256, 0, stream>>>(Wk, Wkp);
  pack_wt_kernel<<<(D_ * D_) / 256, 256, 0, stream>>>(Wv, Wvp);
  pack_wt_kernel<<<(D_ * D_) / 256, 256, 0, stream>>>(Wo, Wop);
  pack_wt_kernel<<<(D_ * D_) / 256, 256, 0, stream>>>(W1, W1p);
  pack_wt_kernel<<<(D_ * D_) / 256, 256, 0, stream>>>(W2, W2p);
  pack_posv_kernel<<<(DH_ * NRELP + 255) / 256, 256, 0, stream>>>(posV, posVp);

  dim3 gg(MTOK / 64, D_ / 128);
  // Q,K -> (B,H,L,DH) bf16 ; V -> (B,H,DH,L) bf16
  gemm_bf16_kernel<<<gg, 128, 0, stream>>>(xb, Wqp, bq, nullptr, Qbf, MTOK, D_, D_, 1, 0);
  gemm_bf16_kernel<<<gg, 128, 0, stream>>>(xb, Wkp, bk, nullptr, Kbf, MTOK, D_, D_, 1, 0);
  gemm_bf16_kernel<<<gg, 128, 0, stream>>>(xb, Wvp, bv, nullptr, Vt,  MTOK, D_, D_, 2, 0);

  qp_kernel<<<BH_ * L_, 160, 0, stream>>>(Qbf, posK, Qp);

  attn_kernel<<<dim3(L_ / 16, BH_), 128, 0, stream>>>(Qbf, Kbf, Vt, Qp, posVp, pm, attnIn);

  // attn = O @ Wo + bo (f32)
  gemm_bf16_kernel<<<gg, 128, 0, stream>>>(attnIn, Wop, bo, attnF, nullptr, MTOK, D_, D_, 0, 0);

  bn_stats_kernel<<<D_ / 256, 256, 0, stream>>>(attnF, mean1, var1);
  bn_apply_kernel<<<(NTOT + 255) / 256, 256, 0, stream>>>(attnF, mean1, var1, ga, bta, x,
                                                          x1F, x1B, 0);

  // ff path
  gemm_bf16_kernel<<<gg, 128, 0, stream>>>(x1B, W1p, b1, nullptr, h1B, MTOK, D_, D_, 0, 1);
  gemm_bf16_kernel<<<gg, 128, 0, stream>>>(h1B, W2p, b2, ffF, nullptr, MTOK, D_, D_, 0, 0);

  bn_stats_kernel<<<D_ / 256, 256, 0, stream>>>(ffF, mean2, var2);
  bn_apply_kernel<<<(NTOT + 255) / 256, 256, 0, stream>>>(ffF, mean2, var2, gf, bfv, x1F,
                                                          out, nullptr, 1);
}
